// MTNN_66271345377548
// MI455X (gfx1250) — compile-verified, matrix operand order checked by reading, not run on hardware
//
#include <hip/hip_runtime.h>
#include <hip/hip_bf16.h>

// ---------------- problem constants (from reference) ----------------
#define B_    512
#define T_    200
#define IS_   64
#define ID_   128
#define HS_   256
#define HD_   256
#define NN_   256
#define DHEAD_ 768   // HS + 2*HD

typedef __attribute__((ext_vector_type(16))) __bf16 v16bf;
typedef __attribute__((ext_vector_type(8)))  float  v8f;

// ------------------------------------------------------------------
// Gather a 16x32 bf16 WMMA operand tile from a row-major matrix.
// A-operand: row0 = m-tile origin, lane(0..15)=row M, K striped per ISA:
//   VGPR v (pair 2v,2v+1): k = k0base + (v>>2)*16 + (lane>>4)*8 + (v&3)*2
// B-operand (B = W^T so B^T = W): identical gather with row0 = n-tile origin.
// ------------------------------------------------------------------
template <typename SrcT>
__device__ inline v16bf load_tile_rowmajor(const SrcT* __restrict__ X, int ld,
                                           int row0, int k0base, int lane) {
    const int m  = row0 + (lane & 15);
    const int lh = lane >> 4;
    const SrcT* p = X + (size_t)m * (size_t)ld;
    v16bf out;
#pragma unroll
    for (int v = 0; v < 8; ++v) {
        const int k0 = k0base + ((v >> 2) * 16) + (lh * 8) + ((v & 3) * 2);
        out[2 * v]     = (__bf16)(float)p[k0];
        out[2 * v + 1] = (__bf16)(float)p[k0 + 1];
    }
    return out;
}

// ------------------------------------------------------------------
// Generic  Y = act(X @ W^T + b1 (+ b2))  with bf16 output.
// One wave per 16x16 output tile; K-loop of v_wmma_f32_16x16x32_bf16.
// X: [M,K] (f32 or bf16), W: [N,K] f32 row-major, Y: [M,N] bf16.
// ------------------------------------------------------------------
template <typename SrcT, bool RELU>
__global__ void __launch_bounds__(32)
gemm_bias_bf16(const SrcT* __restrict__ X, const float* __restrict__ W,
               const float* __restrict__ b1, const float* __restrict__ b2,
               __bf16* __restrict__ Y, int M, int N, int K) {
    const int ntn  = N >> 4;
    const int tile = blockIdx.x;
    const int m0   = (tile / ntn) << 4;
    const int n0   = (tile % ntn) << 4;
    const int lane = threadIdx.x & 31;

    v8f acc = {};
    for (int k0 = 0; k0 < K; k0 += 32) {
        v16bf a = load_tile_rowmajor<SrcT>(X, K, m0, k0, lane);
        v16bf b = load_tile_rowmajor<float>(W, K, n0, k0, lane);
        acc = __builtin_amdgcn_wmma_f32_16x16x32_bf16(
            false, a, false, b, (short)0, acc, false, false);
    }

    const int n  = n0 + (lane & 15);
    const int lh = lane >> 4;
    float bias = (b1 ? b1[n] : 0.0f) + (b2 ? b2[n] : 0.0f);
#pragma unroll
    for (int v = 0; v < 8; ++v) {
        const int m = m0 + v + 8 * lh;
        float val = acc[v] + bias;
        if (RELU) val = fmaxf(val, 0.0f);
        Y[(size_t)m * (size_t)N + n] = (__bf16)val;
    }
}

// ------------------------------------------------------------------
// Bidirectional ReLU-RNN scan:  h_t = relu(x_t + h_{t-1} @ Whh^T)
// grid = (B/16, 2 directions), block = 512 threads = 16 waves.
// Each workgroup owns 16 batch rows; wave w owns N-tile [16*w, 16*w+16).
// Whh B-operand tiles (8 x v16bf per wave) live in registers for all T
// steps; the 16x256 hidden state is exchanged via 8 KB of LDS per step.
// h_t overwrites x_t in global memory (bf16) -> output buffer == input.
// ------------------------------------------------------------------
__global__ void __launch_bounds__(512)
rnn_scan(const float* __restrict__ Whh_fw, const float* __restrict__ Whh_bw,
         __bf16* __restrict__ xf, __bf16* __restrict__ xb) {
    __shared__ __bf16 hbuf[16 * HD_];   // 8 KB: h_{t-1} as bf16, row-major

    const int   dir  = blockIdx.y;
    const float* Whh = dir ? Whh_bw : Whh_fw;
    __bf16*     xio  = dir ? xb : xf;

    const int b0   = blockIdx.x * 16;
    const int wave = threadIdx.x >> 5;   // 0..15: owns one 16-wide N tile
    const int lane = threadIdx.x & 31;
    const int lh   = lane >> 4;
    const int n0   = wave << 4;
    const int n    = n0 + (lane & 15);

    // Preload constant B-operand tiles for all 8 K-slices (64 VGPRs).
    v16bf Bt[8];
#pragma unroll
    for (int kt = 0; kt < 8; ++kt)
        Bt[kt] = load_tile_rowmajor<float>(Whh, HD_, n0, kt * 32, lane);

    // h0 = 0
    for (int i = threadIdx.x; i < 16 * HD_; i += 512) hbuf[i] = (__bf16)0.0f;
    __syncthreads();

    for (int step = 0; step < T_; ++step) {
        const int t = dir ? (T_ - 1 - step) : step;

        // acc = h_{t-1} @ Whh^T   (A tiles streamed from LDS)
        v8f acc = {};
#pragma unroll
        for (int kt = 0; kt < 8; ++kt) {
            const __bf16* hp = &hbuf[(lane & 15) * HD_];
            v16bf a;
#pragma unroll
            for (int v = 0; v < 8; ++v) {
                const int k0 = kt * 32 + ((v >> 2) * 16) + (lh * 8) + ((v & 3) * 2);
                a[2 * v]     = hp[k0];
                a[2 * v + 1] = hp[k0 + 1];
            }
            acc = __builtin_amdgcn_wmma_f32_16x16x32_bf16(
                false, a, false, Bt[kt], (short)0, acc, false, false);
        }

        // h_t = relu(x_t + acc); write back over x_t (bf16).
        float hv[8];
#pragma unroll
        for (int v = 0; v < 8; ++v) {
            const int m = v + 8 * lh;
            const size_t g = ((size_t)(b0 + m) * T_ + t) * HD_ + n;
            const float x = (float)xio[g];
            const float h = fmaxf(x + acc[v], 0.0f);
            hv[v] = h;
            xio[g] = (__bf16)h;
        }

        __syncthreads();   // all waves done reading h_{t-1}
#pragma unroll
        for (int v = 0; v < 8; ++v) {
            const int m = v + 8 * lh;
            hbuf[m * HD_ + n] = (__bf16)hv[v];
        }
        __syncthreads();   // h_t visible for next step
    }
}

// ------------------------------------------------------------------
// Expert head: y[b,t] = relu( [s(b) | hf(b,t) | hb(b,t)] . Wg[b] + bg[b] )
// One wave per (b,t); cross-lane shuffle reduction. OUT==1.
// ------------------------------------------------------------------
__global__ void __launch_bounds__(256)
head_kernel(const __bf16* __restrict__ sbuf, const __bf16* __restrict__ hf,
            const __bf16* __restrict__ hb, const float* __restrict__ W_heads,
            const float* __restrict__ b_heads, const int* __restrict__ order,
            float* __restrict__ out) {
    const int wave = threadIdx.x >> 5;
    const int lane = threadIdx.x & 31;
    const int idx  = blockIdx.x * 8 + wave;          // flat (b,t)
    if (idx >= B_ * T_) return;
    const int b = idx / T_;
    const int t = idx - b * T_;

    const int e = order[b];
    const float*  Wg = W_heads + (size_t)e * DHEAD_;
    const __bf16* sp = sbuf + (size_t)b * HS_;
    const __bf16* fp = hf + ((size_t)b * T_ + t) * HD_;
    const __bf16* bp = hb + ((size_t)b * T_ + t) * HD_;

    float sum = 0.0f;
    for (int j = lane; j < HS_; j += 32) sum += (float)sp[j] * Wg[j];
    for (int j = lane; j < HD_; j += 32) sum += (float)fp[j] * Wg[HS_ + j];
    for (int j = lane; j < HD_; j += 32) sum += (float)bp[j] * Wg[HS_ + HD_ + j];
#pragma unroll
    for (int off = 16; off; off >>= 1) sum += __shfl_xor(sum, off, 32);

    if (lane == 0) out[idx] = fmaxf(sum + b_heads[e], 0.0f);
}

// ------------------------------------------------------------------
extern "C" void kernel_launch(void* const* d_in, const int* in_sizes, int n_in,
                              void* d_out, int out_size, void* d_ws, size_t ws_size,
                              hipStream_t stream) {
    (void)in_sizes; (void)n_in; (void)out_size; (void)ws_size;

    const float* x_static = (const float*)d_in[0];   // [B,IS]
    const float* x_dyn    = (const float*)d_in[1];   // [B,T,ID]
    const int*   order    = (const int*)  d_in[2];   // [B]
    const float* W_static = (const float*)d_in[3];   // [HS,IS]
    const float* b_static = (const float*)d_in[4];
    const float* W_dyn    = (const float*)d_in[5];   // [HD,ID]
    const float* b_dyn    = (const float*)d_in[6];
    const float* Wih_f    = (const float*)d_in[7];   // [HD,HD]
    const float* Whh_f    = (const float*)d_in[8];
    const float* bih_f    = (const float*)d_in[9];
    const float* bhh_f    = (const float*)d_in[10];
    const float* Wih_b    = (const float*)d_in[11];
    const float* Whh_b    = (const float*)d_in[12];
    const float* bih_b    = (const float*)d_in[13];
    const float* bhh_b    = (const float*)d_in[14];
    const float* W_heads  = (const float*)d_in[15];  // [NN,1,DHEAD]
    const float* b_heads  = (const float*)d_in[16];  // [NN,1]
    float* out = (float*)d_out;                      // [B*OUT, T]

    // Workspace layout (bf16 intermediates): d | xf/hf | xb/hb | s
    const size_t BT   = (size_t)B_ * T_;
    const size_t big  = BT * HD_ * sizeof(__bf16);   // 52,428,800 B each
    char* ws = (char*)d_ws;
    __bf16* d_buf  = (__bf16*)(ws);
    __bf16* xf_buf = (__bf16*)(ws + big);            // becomes hf in-place
    __bf16* xb_buf = (__bf16*)(ws + 2 * big);        // becomes hb in-place
    __bf16* s_buf  = (__bf16*)(ws + 3 * big);        // [B,HS]

    // 1) s = relu(x_static @ W_static^T + b_static)          [512x256, K=64]
    gemm_bias_bf16<float, true><<<dim3((B_ / 16) * (HS_ / 16)), 32, 0, stream>>>(
        x_static, W_static, b_static, nullptr, s_buf, B_, HS_, IS_);

    // 2) d = relu(x_dynamic @ W_dyn^T + b_dyn)               [102400x256, K=128]
    gemm_bias_bf16<float, true><<<dim3((BT / 16) * (HD_ / 16)), 32, 0, stream>>>(
        x_dyn, W_dyn, b_dyn, nullptr, d_buf, (int)BT, HD_, ID_);

    // 3) xf = d @ Wih_f^T + (bih_f + bhh_f)                  [102400x256, K=256]
    gemm_bias_bf16<__bf16, false><<<dim3((BT / 16) * (HD_ / 16)), 32, 0, stream>>>(
        d_buf, Wih_f, bih_f, bhh_f, xf_buf, (int)BT, HD_, HD_);

    // 4) xb = d @ Wih_b^T + (bih_b + bhh_b)
    gemm_bias_bf16<__bf16, false><<<dim3((BT / 16) * (HD_ / 16)), 32, 0, stream>>>(
        d_buf, Wih_b, bih_b, bhh_b, xb_buf, (int)BT, HD_, HD_);

    // 5) Bidirectional ReLU-RNN scan (h overwrites x in place)
    rnn_scan<<<dim3(B_ / 16, 2), 512, 0, stream>>>(Whh_f, Whh_b, xf_buf, xb_buf);

    // 6) Expert head -> out[b*T + t]
    head_kernel<<<dim3((B_ * T_) / 8), 256, 0, stream>>>(
        s_buf, xf_buf, xb_buf, W_heads, b_heads, order, out);
}
